// ComplexMultiheadAttention_9019431322190
// MI455X (gfx1250) — matrix operand order repaired and checked by reading, not compiled, not run on hardware
//
#include <hip/hip_runtime.h>
#include <hip/hip_bf16.h>

typedef __attribute__((ext_vector_type(16))) _Float16 v16h;
typedef __attribute__((ext_vector_type(8)))  _Float16 v8h;
typedef __attribute__((ext_vector_type(8)))  float    v8f;

#define CB 2
#define CT 2048
#define CD 1024
#define CH 16
#define CDH 64

// ---------------------------------------------------------------------------
// WMMA helper: D = A(16x32 f16) * B(32x16 f16) + C(16x16 f32)
// ---------------------------------------------------------------------------
__device__ __forceinline__ v8f wmma16(v16h a, v16h b, v8f c) {
  return __builtin_amdgcn_wmma_f32_16x16x32_f16(
      /*neg_a=*/false, a, /*neg_b=*/false, b,
      /*c_mod=*/(short)0, c, /*reuse_a=*/false, /*reuse_b=*/false);
}

// A-fragment (16x32, row-major source, ld in halfs).
// ISA: lanes 0-15 -> M=lane, halfs {K0..K0+7, K0+16..K0+23};
//      lanes 16-31 -> M=lane-16, halfs {K0+8..K0+15, K0+24..K0+31}.
__device__ __forceinline__ v16h load_a16(const _Float16* __restrict__ p,
                                         int ld, int k0, int lane) {
  const int m = lane & 15;
  const int k = k0 + ((lane >> 4) << 3);
  const _Float16* q = p + (size_t)m * ld + k;
  v8h lo = *(const v8h*)q;
  v8h hi = *(const v8h*)(q + 16);
  v16h a;
#pragma unroll
  for (int i = 0; i < 8; ++i) { a[i] = lo[i]; a[i + 8] = hi[i]; }
  return a;
}

// B-fragment (32x16). Source is [N,K] row-major, ld in halfs.
// ISA: lanes 0-15 -> N=lane, K={k0..k0+7, k0+16..k0+23}; lanes 16-31 -> K={k0+8..k0+15, k0+24..k0+31}.
__device__ __forceinline__ v16h load_b16(const _Float16* __restrict__ p,
                                         int ld, int k0, int lane) {
  const int n = lane & 15;
  const int k = k0 + ((lane >> 4) << 3);
  const _Float16* q = p + (size_t)n * ld + k;
  v8h lo = *(const v8h*)q;
  v8h hi = *(const v8h*)(q + 16);
  v16h b;
#pragma unroll
  for (int i = 0; i < 8; ++i) { b[i] = lo[i]; b[i + 8] = hi[i]; }
  return b;
}

// ---------------------------------------------------------------------------
// gfx1250 async copy: global -> LDS, 16 bytes per lane, tracked by ASYNCcnt.
// LDS destination offset = low 32 bits of the generic pointer (aperture rule).
// ---------------------------------------------------------------------------
__device__ __forceinline__ unsigned lds_off_u32(const void* p) {
  return (unsigned)(size_t)p;
}
__device__ __forceinline__ void async_copy_b128(unsigned ldsOff, const void* g) {
  asm volatile("global_load_async_to_lds_b128 %0, %1, off"
               :: "v"(ldsOff), "v"(g) : "memory");
}
__device__ __forceinline__ void wait_async0() {
  asm volatile("s_wait_asynccnt 0x0" ::: "memory");
}

// ---------------------------------------------------------------------------
// fp32 -> fp16 convert
// ---------------------------------------------------------------------------
__global__ void conv_f32_f16(const float* __restrict__ in,
                             _Float16* __restrict__ out, int n) {
  int i = blockIdx.x * blockDim.x + threadIdx.x;
  if (i < n) out[i] = (_Float16)in[i];
}

// W [K,N] f32 -> Wt [N,K] f16  (so B-fragments become contiguous loads)
__global__ void transpose_w(const float* __restrict__ in,
                            _Float16* __restrict__ out, int K, int N) {
  int i = blockIdx.x * blockDim.x + threadIdx.x;
  if (i < K * N) {
    int k = i / N, n = i % N;
    out[(size_t)n * K + k] = (_Float16)in[i];
  }
}

// ---------------------------------------------------------------------------
// Complex GEMM:  Yr = Xr*Wr - Xi*Wi + br ;  Yi = Xi*Wr + Xr*Wi + bi
// MODE 0: f16 [M,N] row-major (Q,K) | MODE 1: f16 [B,H,d,T] (V) | MODE 2: f32
// Block = 128 threads (4 waves); each wave owns a 16(M) x 64(N) tile.
// ---------------------------------------------------------------------------
template<int MODE>
__global__ __launch_bounds__(128) void cgemm_wmma(
    const _Float16* __restrict__ Xr, const _Float16* __restrict__ Xi,
    const _Float16* __restrict__ Wtr, const _Float16* __restrict__ Wti,
    const float* __restrict__ br, const float* __restrict__ bi,
    void* __restrict__ outR_, void* __restrict__ outI_,
    int M, int N, int K) {
  const int lane  = threadIdx.x & 31;
  const int wave  = threadIdx.x >> 5;
  const int mtile = blockIdx.x * 64 + wave * 16;
  const int ntile = blockIdx.y * 64;

  v8f accR[4] = {};  // Xr*Wr
  v8f accS[4] = {};  // Xi*Wi (subtracted in epilogue; f16 WMMA NEG[1:0] must be 0)
  v8f accI[4] = {};  // Xi*Wr + Xr*Wi

  const _Float16* xr = Xr + (size_t)mtile * K;
  const _Float16* xi = Xi + (size_t)mtile * K;

  for (int k0 = 0; k0 < K; k0 += 32) {
    v16h ar = load_a16(xr, K, k0, lane);
    v16h ai = load_a16(xi, K, k0, lane);
#pragma unroll
    for (int nt = 0; nt < 4; ++nt) {
      const _Float16* wrp = Wtr + (size_t)(ntile + nt * 16) * K;
      const _Float16* wip = Wti + (size_t)(ntile + nt * 16) * K;
      v16h bwr = load_b16(wrp, K, k0, lane);
      v16h bwi = load_b16(wip, K, k0, lane);
      accR[nt] = wmma16(ar, bwr, accR[nt]);
      accS[nt] = wmma16(ai, bwi, accS[nt]);
      accI[nt] = wmma16(ai, bwr, accI[nt]);
      accI[nt] = wmma16(ar, bwi, accI[nt]);
    }
  }

  const int lane_n = lane & 15;
  const int mbase  = (lane >> 4) * 8;
#pragma unroll
  for (int nt = 0; nt < 4; ++nt) {
    const int n = ntile + nt * 16 + lane_n;
    const float vbr = br[n], vbi = bi[n];
#pragma unroll
    for (int r = 0; r < 8; ++r) {
      const int m = mtile + mbase + r;
      const float vr = accR[nt][r] - accS[nt][r] + vbr;
      const float vi = accI[nt][r] + vbi;
      if (MODE == 0) {
        ((_Float16*)outR_)[(size_t)m * N + n] = (_Float16)vr;
        ((_Float16*)outI_)[(size_t)m * N + n] = (_Float16)vi;
      } else if (MODE == 1) {
        const int bb = m >> 11, t = m & (CT - 1);
        const int hh = n >> 6,  f = n & (CDH - 1);
        const size_t idx = (((size_t)bb * CH + hh) * CDH + f) * CT + t;
        ((_Float16*)outR_)[idx] = (_Float16)vr;
        ((_Float16*)outI_)[idx] = (_Float16)vi;
      } else {
        ((float*)outR_)[(size_t)m * N + n] = vr;
        ((float*)outI_)[(size_t)m * N + n] = vi;
      }
    }
  }
}

// ---------------------------------------------------------------------------
// Flash attention: scores = (qr.kr + qi.ki)/8, softmax, out = P@V (r and i).
// Block = 4 waves = 64 query rows of one (b,h). K/V tiles (32 keys) are
// staged global->LDS with async b128 copies, double-buffered so the copy for
// step j+1 overlaps compute of step j. Row sums of P come from an extra
// WMMA against an all-ones B (replaces 32 bpermute+wait chains per step).
// ---------------------------------------------------------------------------
__global__ __launch_bounds__(128) void attn_wmma(
    const _Float16* __restrict__ Qr, const _Float16* __restrict__ Qi,
    const _Float16* __restrict__ Kr, const _Float16* __restrict__ Ki,
    const _Float16* __restrict__ Vtr, const _Float16* __restrict__ Vti,
    _Float16* __restrict__ Or, _Float16* __restrict__ Oi) {
  __shared__ _Float16 lsK[2][2][32 * CDH];  // [buf][r/i][key][feat]
  __shared__ _Float16 lsV[2][2][CDH * 32];  // [buf][r/i][feat][key]
  __shared__ _Float16 lsP[4][16 * 32];      // per-wave P tile (C -> A layout)

  const int tid  = threadIdx.x;
  const int lane = tid & 31;
  const int wave = tid >> 5;
  const int blk  = blockIdx.x;
  const int tile = blk & 31;   // 64-row query tile (T/64 = 32)
  const int bh   = blk >> 5;
  const int b    = bh >> 4;
  const int h    = bh & 15;

  const _Float16* krb = Kr  + (size_t)b * CT * CD + h * CDH;
  const _Float16* kib = Ki  + (size_t)b * CT * CD + h * CDH;
  const _Float16* vtr = Vtr + (size_t)bh * CDH * CT;  // [d, T] per (b,h)
  const _Float16* vti = Vti + (size_t)bh * CDH * CT;

  // stage one 32-key step (K r/i + V r/i tiles) into LDS buffer `buf`
  auto stage = [&](int j0, int buf) {
    const unsigned kr_o = lds_off_u32(&lsK[buf][0][0]);
    const unsigned ki_o = lds_off_u32(&lsK[buf][1][0]);
    const unsigned vr_o = lds_off_u32(&lsV[buf][0][0]);
    const unsigned vi_o = lds_off_u32(&lsV[buf][1][0]);
#pragma unroll
    for (int c = tid; c < 256; c += 128) {        // K: 32 keys x 128B
      const int key = c >> 3, x = (c & 7) * 16;
      const size_t gb = ((size_t)(j0 + key) * CD) * 2 + x;
      async_copy_b128(kr_o + c * 16, (const char*)krb + gb);
      async_copy_b128(ki_o + c * 16, (const char*)kib + gb);
    }
#pragma unroll
    for (int c = tid; c < 256; c += 128) {        // V: 64 feats x 64B
      const int f = c >> 2, x = (c & 3) * 16;
      const size_t gb = ((size_t)f * CT + j0) * 2 + x;
      async_copy_b128(vr_o + c * 16, (const char*)vtr + gb);
      async_copy_b128(vi_o + c * 16, (const char*)vti + gb);
    }
  };

  // Q fragments for this wave's 16 rows
  const size_t qoff = ((size_t)b * CT + tile * 64 + wave * 16) * CD + h * CDH;
  v16h aqr[2], aqi[2];
  aqr[0] = load_a16(Qr + qoff, CD, 0,  lane);
  aqr[1] = load_a16(Qr + qoff, CD, 32, lane);
  aqi[0] = load_a16(Qi + qoff, CD, 0,  lane);
  aqi[1] = load_a16(Qi + qoff, CD, 32, lane);

  v16h vones;
#pragma unroll
  for (int i = 0; i < 16; ++i) vones[i] = (_Float16)1.0f;

  v8f aor[4] = {}, aoi[4] = {};
  float m8[8], l8[8];
#pragma unroll
  for (int r = 0; r < 8; ++r) { m8[r] = -1e30f; l8[r] = 0.f; }

  const int lane_n = lane & 15;
  const int mbase  = (lane >> 4) * 8;
  _Float16* pw = &lsP[wave][0];

  stage(0, 0);  // prologue: prefetch first step

  for (int j0 = 0; j0 < CT; j0 += 32) {
    const int cur = (j0 >> 5) & 1;
    wait_async0();       // my copies for `cur` complete
    __syncthreads();     // everyone's copies complete; prev compute done
    if (j0 + 32 < CT) stage(j0 + 32, cur ^ 1);  // overlap next copy w/ compute

    // ---- scores: two 16-key subtiles from LDS K tiles ----
    const _Float16* kr0 = &lsK[cur][0][0];
    const _Float16* ki0 = &lsK[cur][1][0];
    v8f s0 = {}, s1 = {};
#pragma unroll
    for (int kc = 0; kc < 2; ++kc) {
      s0 = wmma16(aqr[kc], load_b16(kr0,            CDH, kc * 32, lane), s0);
      s0 = wmma16(aqi[kc], load_b16(ki0,            CDH, kc * 32, lane), s0);
      s1 = wmma16(aqr[kc], load_b16(kr0 + 16 * CDH, CDH, kc * 32, lane), s1);
      s1 = wmma16(aqi[kc], load_b16(ki0 + 16 * CDH, CDH, kc * 32, lane), s1);
    }

    // ---- online softmax: shuffle reduction for row max only ----
    float p0[8], p1[8], corr[8];
#pragma unroll
    for (int r = 0; r < 8; ++r) {
      const float a0 = s0[r] * 0.125f;   // 1/sqrt(64)
      const float a1 = s1[r] * 0.125f;
      float mx = fmaxf(a0, a1);
#pragma unroll
      for (int msk = 1; msk < 16; msk <<= 1)
        mx = fmaxf(mx, __shfl_xor(mx, msk, 32));
      const float mnew = fmaxf(m8[r], mx);
      corr[r] = __expf(m8[r] - mnew);
      p0[r]   = __expf(a0 - mnew);
      p1[r]   = __expf(a1 - mnew);
      m8[r]   = mnew;
#pragma unroll
      for (int nt = 0; nt < 4; ++nt) { aor[nt][r] *= corr[r]; aoi[nt][r] *= corr[r]; }
    }

    // ---- P: C-layout -> per-wave LDS -> A-fragment ----
#pragma unroll
    for (int r = 0; r < 8; ++r) {
      pw[(mbase + r) * 32 + lane_n]      = (_Float16)p0[r];
      pw[(mbase + r) * 32 + 16 + lane_n] = (_Float16)p1[r];
    }
    v16h pa = load_a16(pw, 32, 0, lane);

    // row sums of P via WMMA against all-ones B (lands in C-fragment slots)
    v8f zsum = {};
    v8f lsum = wmma16(pa, vones, zsum);
#pragma unroll
    for (int r = 0; r < 8; ++r) l8[r] = l8[r] * corr[r] + lsum[r];

    // ---- out += P @ V from LDS V tiles ([feat][key], ld=32) ----
    const _Float16* vr0 = &lsV[cur][0][0];
    const _Float16* vi0 = &lsV[cur][1][0];
#pragma unroll
    for (int nt = 0; nt < 4; ++nt) {
      aor[nt] = wmma16(pa, load_b16(vr0 + nt * 16 * 32, 32, 0, lane), aor[nt]);
      aoi[nt] = wmma16(pa, load_b16(vi0 + nt * 16 * 32, 32, 0, lane), aoi[nt]);
    }
  }

  // ---- normalize, store [B,T,D] f16 (input layout for O projection) ----
#pragma unroll
  for (int r = 0; r < 8; ++r) {
    const float inv = 1.0f / l8[r];
    const int t = tile * 64 + wave * 16 + mbase + r;
    const size_t rowbase = ((size_t)b * CT + t) * CD + h * CDH;
#pragma unroll
    for (int nt = 0; nt < 4; ++nt) {
      Or[rowbase + nt * 16 + lane_n] = (_Float16)(aor[nt][r] * inv);
      Oi[rowbase + nt * 16 + lane_n] = (_Float16)(aoi[nt][r] * inv);
    }
  }
}

// ---------------------------------------------------------------------------
extern "C" void kernel_launch(void* const* d_in, const int* in_sizes, int n_in,
                              void* d_out, int out_size, void* d_ws, size_t ws_size,
                              hipStream_t stream) {
  (void)in_sizes; (void)n_in; (void)out_size; (void)ws_size;

  const float* x_real = (const float*)d_in[0];
  const float* x_imag = (const float*)d_in[1];
  const float* wsrc[8] = {
    (const float*)d_in[2],  (const float*)d_in[3],   // q_wr, q_wi
    (const float*)d_in[6],  (const float*)d_in[7],   // k_wr, k_wi
    (const float*)d_in[10], (const float*)d_in[11],  // v_wr, v_wi
    (const float*)d_in[14], (const float*)d_in[15],  // o_wr, o_wi
  };
  const float* q_br = (const float*)d_in[4];  const float* q_bi = (const float*)d_in[5];
  const float* k_br = (const float*)d_in[8];  const float* k_bi = (const float*)d_in[9];
  const float* v_br = (const float*)d_in[12]; const float* v_bi = (const float*)d_in[13];
  const float* o_br = (const float*)d_in[16]; const float* o_bi = (const float*)d_in[17];

  const int    M   = CB * CT;            // 4096
  const size_t XSZ = (size_t)M * CD;     // 4M elements
  const size_t WSZ = (size_t)CD * CD;    // 1M elements

  char* ws = (char*)d_ws;
  size_t off = 0;
  auto alloc_h = [&](size_t elems) {
    _Float16* p = (_Float16*)(ws + off);
    off += elems * sizeof(_Float16);
    return p;
  };
  _Float16* Xr16 = alloc_h(XSZ);
  _Float16* Xi16 = alloc_h(XSZ);
  _Float16* Wt[8];
  for (int i = 0; i < 8; ++i) Wt[i] = alloc_h(WSZ);
  _Float16* Qr  = alloc_h(XSZ);
  _Float16* Qi  = alloc_h(XSZ);
  _Float16* Kr_ = alloc_h(XSZ);
  _Float16* Ki_ = alloc_h(XSZ);
  _Float16* Vtr = alloc_h(XSZ);   // [B,H,d,T]
  _Float16* Vti = alloc_h(XSZ);
  _Float16* Or16 = alloc_h(XSZ);
  _Float16* Oi16 = alloc_h(XSZ);

  const int thr = 256;
  conv_f32_f16<<<(int)((XSZ + thr - 1) / thr), thr, 0, stream>>>(x_real, Xr16, (int)XSZ);
  conv_f32_f16<<<(int)((XSZ + thr - 1) / thr), thr, 0, stream>>>(x_imag, Xi16, (int)XSZ);
  for (int i = 0; i < 8; ++i)
    transpose_w<<<(int)((WSZ + thr - 1) / thr), thr, 0, stream>>>(wsrc[i], Wt[i], CD, CD);

  dim3 gg(M / 64, CD / 64);
  cgemm_wmma<0><<<gg, 128, 0, stream>>>(Xr16, Xi16, Wt[0], Wt[1], q_br, q_bi, Qr,  Qi,  M, CD, CD);
  cgemm_wmma<0><<<gg, 128, 0, stream>>>(Xr16, Xi16, Wt[2], Wt[3], k_br, k_bi, Kr_, Ki_, M, CD, CD);
  cgemm_wmma<1><<<gg, 128, 0, stream>>>(Xr16, Xi16, Wt[4], Wt[5], v_br, v_bi, Vtr, Vti, M, CD, CD);

  attn_wmma<<<CB * CH * (CT / 64), 128, 0, stream>>>(Qr, Qi, Kr_, Ki_, Vtr, Vti, Or16, Oi16);

  float* yr = (float*)d_out;
  float* yi = yr + (size_t)M * CD;
  cgemm_wmma<2><<<gg, 128, 0, stream>>>(Or16, Oi16, Wt[6], Wt[7], o_br, o_bi, yr, yi, M, CD, CD);
}
